// MAB_5669356835173
// MI455X (gfx1250) — matrix operand order repaired and checked by reading, not compiled, not run on hardware
//
#include <hip/hip_runtime.h>

// ---------------------------------------------------------------------------
// MAB (multi-head attention block) for MI455X / gfx1250 (wave32, WMMA).
// B=8, NQ=NK=1024, D=1024, H=16, HD=64.
// Matmuls in bf16 via v_wmma_f32_16x16x32_bf16 (f32 accumulate).
// GEMM tiles staged with GLOBAL_LOAD_ASYNC_TO_LDS_B128, double-buffered LDS.
// ---------------------------------------------------------------------------

typedef __attribute__((ext_vector_type(8)))  float  v8f;
typedef __attribute__((ext_vector_type(16))) __bf16 v16bf;

union FragBF {
    uint4          u[2];
    unsigned short s[16];
    v16bf          v;
};

union Pack8 {
    unsigned short s[8];
    uint4          u;
};

static __device__ __forceinline__ unsigned short f2bf(float f) {
    unsigned int u = __float_as_uint(f);
    u += 0x7FFFu + ((u >> 16) & 1u);   // round-to-nearest-even
    return (unsigned short)(u >> 16);
}

static __device__ __forceinline__ v8f wmma_bf16(const FragBF& a, const FragBF& b, v8f c) {
    return __builtin_amdgcn_wmma_f32_16x16x32_bf16(
        false, a.v, false, b.v, (short)0, c, false, false);
}

// per-lane 16-byte async copy: global(base + goff bytes) -> LDS(ldsoff bytes)
#define ASYNC_B128(ldsoff, goff, base)                                 \
    asm volatile("global_load_async_to_lds_b128 %0, %1, %2"            \
                 :: "v"(ldsoff), "v"(goff), "s"(base) : "memory")

#define BB   8
#define NQ_  1024
#define NK_  1024
#define DV_  1024
#define H_   16
#define HD_  64

// ---------------------------------------------------------------------------
// Weight transpose + f32->bf16:  Wt[n][k] = bf16(W[k][n]),  1024x1024.
// grid (32,32), block 256 (32x8).
// ---------------------------------------------------------------------------
__global__ __launch_bounds__(256) void wtrans_kernel(
    const float* __restrict__ W, unsigned short* __restrict__ Wt)
{
    __shared__ float t[32][33];
    const int tx = threadIdx.x & 31;
    const int ty = threadIdx.x >> 5;
    const int n0 = blockIdx.x * 32;
    const int k0 = blockIdx.y * 32;
    #pragma unroll
    for (int i = 0; i < 4; ++i)
        t[ty + i * 8][tx] = W[(size_t)(k0 + ty + i * 8) * 1024 + n0 + tx];
    __syncthreads();
    #pragma unroll
    for (int i = 0; i < 4; ++i)
        Wt[(size_t)(n0 + ty + i * 8) * 1024 + k0 + tx] = f2bf(t[tx][ty + i * 8]);
}

// ---------------------------------------------------------------------------
// f32 -> bf16 convert, 8 elements / thread.
// ---------------------------------------------------------------------------
__global__ __launch_bounds__(256) void cvt_kernel(
    const float* __restrict__ in, unsigned short* __restrict__ out)
{
    const size_t i = ((size_t)blockIdx.x * 256 + threadIdx.x) * 8;
    const float4 a = *(const float4*)(in + i);
    const float4 b = *(const float4*)(in + i + 4);
    Pack8 p;
    p.s[0] = f2bf(a.x); p.s[1] = f2bf(a.y); p.s[2] = f2bf(a.z); p.s[3] = f2bf(a.w);
    p.s[4] = f2bf(b.x); p.s[5] = f2bf(b.y); p.s[6] = f2bf(b.z); p.s[7] = f2bf(b.w);
    *(uint4*)(out + i) = p.u;
}

// ---------------------------------------------------------------------------
// GEMM: C[M=8192,N=1024] = A[M,1024]bf16 @ Wt[N,1024]bf16^T (+bias epilogue).
// Double-buffered LDS, tiles staged via async-LDS copies.
// mode 0: out f32                       (Q projection)
// mode 1: out bf16                      (K projection)
// mode 2: out bf16 transposed per head  (V projection -> Vt[b][h][d][nk])
// mode 3: out f32 = resid + relu(v+b)   (output projection)
// ---------------------------------------------------------------------------
struct GemmLds {
    unsigned short As[128 * 40];   // A tile [m][k], pad 40
    unsigned short Bs[64 * 40];    // W tile [n][k], pad 40
};

__global__ __launch_bounds__(256) void gemm_kernel(
    const unsigned short* __restrict__ A, const unsigned short* __restrict__ Wt,
    const float* __restrict__ bias, const float* __restrict__ resid,
    void* __restrict__ outp, int mode)
{
    __shared__ GemmLds sh[2];

    const int tid  = threadIdx.x;
    const int lane = tid & 31;
    const int w    = tid >> 5;
    const int lm   = lane & 15;
    const int kh   = lane >> 4;

    const int m0blk = blockIdx.y * 128;
    const int n0blk = blockIdx.x * 64;
    const int mw = (w >> 1) * 32;
    const int nw = (w & 1) * 32;

    // staging chunk coordinates (16B = 8 bf16 per lane per async op)
    const int rowA0 = tid >> 2,           cA0 = (tid & 3) * 8;          // A chunk 0
    const int rowA1 = (tid + 256) >> 2,   cA1 = (tid & 3) * 8;          // A chunk 1
    const int rowB  = tid >> 2,           cB  = (tid & 3) * 8;          // B chunk

    const unsigned gA0 = ((unsigned)(m0blk + rowA0) * 1024u + cA0) * 2u;
    const unsigned gA1 = ((unsigned)(m0blk + rowA1) * 1024u + cA1) * 2u;
    const unsigned gB  = ((unsigned)(n0blk + rowB)  * 1024u + cB)  * 2u;

    const unsigned ldsA = (unsigned)(size_t)(&sh[0].As[rowA0 * 40 + cA0]);
    const unsigned ldsA1b = (unsigned)(size_t)(&sh[0].As[rowA1 * 40 + cA1]);
    const unsigned ldsB = (unsigned)(size_t)(&sh[0].Bs[rowB * 40 + cB]);
    const unsigned stageBytes = (unsigned)sizeof(GemmLds);

    v8f acc[2][2] = {};

    // prologue: stage 0, kk = 0
    ASYNC_B128(ldsA,          gA0, A);
    ASYNC_B128(ldsA1b,        gA1, A);
    ASYNC_B128(ldsB,          gB,  Wt);

    for (int s = 0; s < 32; ++s) {
        const int p = s & 1;
        if (s + 1 < 32) {
            const unsigned kb = (unsigned)(s + 1) * 64u;  // kk*2 bytes, kk=(s+1)*32
            const unsigned sb = (unsigned)(p ^ 1) * stageBytes;
            ASYNC_B128(ldsA + sb,   gA0 + kb, A);
            ASYNC_B128(ldsA1b + sb, gA1 + kb, A);
            ASYNC_B128(ldsB + sb,   gB + kb,  Wt);
            asm volatile("s_wait_asynccnt 0x3" ::: "memory");
        } else {
            asm volatile("s_wait_asynccnt 0x0" ::: "memory");
        }
        __syncthreads();

        const unsigned short* As = sh[p].As;
        const unsigned short* Bs = sh[p].Bs;
        FragBF a[2], bf[2];
        #pragma unroll
        for (int i = 0; i < 2; ++i) {
            const unsigned short* pp = As + (mw + i * 16 + lm) * 40 + kh * 8;
            a[i].u[0] = *(const uint4*)(pp);
            a[i].u[1] = *(const uint4*)(pp + 16);
        }
        #pragma unroll
        for (int j = 0; j < 2; ++j) {
            const unsigned short* pp = Bs + (nw + j * 16 + lm) * 40 + kh * 8;
            bf[j].u[0] = *(const uint4*)(pp);
            bf[j].u[1] = *(const uint4*)(pp + 16);
        }
        #pragma unroll
        for (int i = 0; i < 2; ++i)
            #pragma unroll
            for (int j = 0; j < 2; ++j)
                acc[i][j] = wmma_bf16(a[i], bf[j], acc[i][j]);
        __syncthreads();
    }

    // epilogue
    #pragma unroll
    for (int i = 0; i < 2; ++i) {
        #pragma unroll
        for (int j = 0; j < 2; ++j) {
            #pragma unroll
            for (int r = 0; r < 8; ++r) {
                int mg = m0blk + mw + i * 16 + kh * 8 + r;
                int ng = n0blk + nw + j * 16 + lm;
                float v = acc[i][j][r] + bias[ng];
                if (mode == 0) {
                    ((float*)outp)[(size_t)mg * 1024 + ng] = v;
                } else if (mode == 1) {
                    ((unsigned short*)outp)[(size_t)mg * 1024 + ng] = f2bf(v);
                } else if (mode == 2) {
                    int b = mg >> 10, nk = mg & 1023;
                    int h = ng >> 6,  d  = ng & 63;
                    ((unsigned short*)outp)[(((size_t)(b * H_ + h) * HD_ + d) << 10) + nk] = f2bf(v);
                } else {
                    float x = resid[(size_t)mg * 1024 + ng];
                    ((float*)outp)[(size_t)mg * 1024 + ng] = x + fmaxf(v, 0.0f);
                }
            }
        }
    }
}

// ---------------------------------------------------------------------------
// Attention: O1 = Qh + softmax(Qh Kh^T / 32) Vh, flash-style online softmax.
// grid (B*H, NQ/128), 256 threads = 8 waves; wave -> one 16-row q tile.
// ---------------------------------------------------------------------------
__global__ __launch_bounds__(256) void attn_kernel(
    const float* __restrict__ Qp, const unsigned short* __restrict__ Kbf,
    const unsigned short* __restrict__ Vt, float* __restrict__ O1)
{
    __shared__ unsigned short Pt[8][16 * 40];   // per-wave P transpose staging

    const int lane = threadIdx.x & 31;
    const int w    = threadIdx.x >> 5;
    const int lm   = lane & 15;
    const int kh   = lane >> 4;

    const int b  = blockIdx.x >> 4;
    const int h  = blockIdx.x & 15;
    const int q0 = (blockIdx.y * 8 + w) * 16;

    const float scale = 0.03125f;   // 1/sqrt(1024)
    FragBF qf[2];
    const float* qrow = Qp + ((size_t)(b * NQ_ + q0 + lm)) * 1024 + h * 64;
    #pragma unroll
    for (int c = 0; c < 2; ++c) {
        #pragma unroll
        for (int e = 0; e < 8; ++e) {
            qf[c].s[e]     = f2bf(qrow[c * 32 + kh * 8 + e] * scale);
            qf[c].s[8 + e] = f2bf(qrow[c * 32 + 16 + kh * 8 + e] * scale);
        }
    }

    v8f o[4] = {};
    float mr[8], lr[8];
    #pragma unroll
    for (int r = 0; r < 8; ++r) { mr[r] = -1e30f; lr[r] = 0.0f; }

    unsigned short* myP = Pt[w];

    for (int kt = 0; kt < NK_; kt += 32) {
        FragBF kb0[2], kb1[2];
        {
            const unsigned short* kr0 =
                Kbf + ((size_t)(b * NK_ + kt + lm)) * 1024 + h * 64 + kh * 8;
            kb0[0].u[0] = *(const uint4*)(kr0);
            kb0[0].u[1] = *(const uint4*)(kr0 + 16);
            kb0[1].u[0] = *(const uint4*)(kr0 + 32);
            kb0[1].u[1] = *(const uint4*)(kr0 + 48);
            const unsigned short* kr1 =
                Kbf + ((size_t)(b * NK_ + kt + 16 + lm)) * 1024 + h * 64 + kh * 8;
            kb1[0].u[0] = *(const uint4*)(kr1);
            kb1[0].u[1] = *(const uint4*)(kr1 + 16);
            kb1[1].u[0] = *(const uint4*)(kr1 + 32);
            kb1[1].u[1] = *(const uint4*)(kr1 + 48);
        }
        v8f s0 = {}, s1 = {};
        s0 = wmma_bf16(qf[0], kb0[0], s0);
        s0 = wmma_bf16(qf[1], kb0[1], s0);
        s1 = wmma_bf16(qf[0], kb1[0], s1);
        s1 = wmma_bf16(qf[1], kb1[1], s1);

        #pragma unroll
        for (int r = 0; r < 8; ++r) {
            float v0 = s0[r], v1 = s1[r];
            float mx = fmaxf(v0, v1);
            mx = fmaxf(mx, __shfl_xor(mx, 1, 32));
            mx = fmaxf(mx, __shfl_xor(mx, 2, 32));
            mx = fmaxf(mx, __shfl_xor(mx, 4, 32));
            mx = fmaxf(mx, __shfl_xor(mx, 8, 32));
            float nm = fmaxf(mr[r], mx);
            float al = __expf(mr[r] - nm);
            float p0 = __expf(v0 - nm);
            float p1 = __expf(v1 - nm);
            float ps = p0 + p1;
            ps += __shfl_xor(ps, 1, 32);
            ps += __shfl_xor(ps, 2, 32);
            ps += __shfl_xor(ps, 4, 32);
            ps += __shfl_xor(ps, 8, 32);
            lr[r] = lr[r] * al + ps;
            mr[r] = nm;
            o[0][r] *= al; o[1][r] *= al; o[2][r] *= al; o[3][r] *= al;
            int rowi = kh * 8 + r;
            myP[rowi * 40 + lm]      = f2bf(p0);
            myP[rowi * 40 + 16 + lm] = f2bf(p1);
        }
        asm volatile("s_wait_dscnt 0" ::: "memory");

        FragBF pf;
        {
            const unsigned short* p = myP + lm * 40 + kh * 8;
            pf.u[0] = *(const uint4*)(p);
            pf.u[1] = *(const uint4*)(p + 16);
        }
        #pragma unroll
        for (int t = 0; t < 4; ++t) {
            int d = t * 16 + lm;
            const unsigned short* vr =
                Vt + (((size_t)(b * H_ + h) * HD_ + d) << 10) + kt + kh * 8;
            FragBF vf;
            vf.u[0] = *(const uint4*)(vr);
            vf.u[1] = *(const uint4*)(vr + 16);
            o[t] = wmma_bf16(pf, vf, o[t]);
        }
    }

    #pragma unroll
    for (int t = 0; t < 4; ++t) {
        #pragma unroll
        for (int r = 0; r < 8; ++r) {
            int m = q0 + kh * 8 + r;
            int d = t * 16 + lm;
            size_t idx = ((size_t)(b * NQ_ + m)) * 1024 + h * 64 + d;
            O1[idx] = o[t][r] / lr[r] + Qp[idx];
        }
    }
}

// ---------------------------------------------------------------------------
// LayerNorm over last dim (1024), one block per row; optional bf16 mirror.
// ---------------------------------------------------------------------------
__global__ __launch_bounds__(256) void ln_kernel(
    const float* __restrict__ in, float* __restrict__ out,
    unsigned short* __restrict__ out_bf,
    const float* __restrict__ g, const float* __restrict__ bet)
{
    __shared__ float s1[256], s2[256];
    const int row = blockIdx.x;
    const int t   = threadIdx.x;
    const int c   = t * 4;
    const float4 x = *(const float4*)(in + (size_t)row * 1024 + c);
    s1[t] = x.x + x.y + x.z + x.w;
    s2[t] = x.x * x.x + x.y * x.y + x.z * x.z + x.w * x.w;
    __syncthreads();
    for (int o2 = 128; o2 > 0; o2 >>= 1) {
        if (t < o2) { s1[t] += s1[t + o2]; s2[t] += s2[t + o2]; }
        __syncthreads();
    }
    const float mean = s1[0] * (1.0f / 1024.0f);
    const float var  = s2[0] * (1.0f / 1024.0f) - mean * mean;
    const float rs   = rsqrtf(var + 1e-5f);
    float4 y;
    y.x = (x.x - mean) * rs * g[c + 0] + bet[c + 0];
    y.y = (x.y - mean) * rs * g[c + 1] + bet[c + 1];
    y.z = (x.z - mean) * rs * g[c + 2] + bet[c + 2];
    y.w = (x.w - mean) * rs * g[c + 3] + bet[c + 3];
    *(float4*)(out + (size_t)row * 1024 + c) = y;
    if (out_bf) {
        unsigned short* ob = out_bf + (size_t)row * 1024 + c;
        ob[0] = f2bf(y.x); ob[1] = f2bf(y.y); ob[2] = f2bf(y.z); ob[3] = f2bf(y.w);
    }
}

// ---------------------------------------------------------------------------
extern "C" void kernel_launch(void* const* d_in, const int* in_sizes, int n_in,
                              void* d_out, int out_size, void* d_ws, size_t ws_size,
                              hipStream_t stream) {
    (void)in_sizes; (void)n_in; (void)out_size; (void)ws_size;
    const float* Q   = (const float*)d_in[0];
    const float* Kin = (const float*)d_in[1];
    const float* Wq  = (const float*)d_in[2];
    const float* bq  = (const float*)d_in[3];
    const float* Wk  = (const float*)d_in[4];
    const float* bk  = (const float*)d_in[5];
    const float* Wv  = (const float*)d_in[6];
    const float* bv  = (const float*)d_in[7];
    const float* Wo  = (const float*)d_in[8];
    const float* bo  = (const float*)d_in[9];
    const float* g0  = (const float*)d_in[10];
    const float* b0  = (const float*)d_in[11];
    const float* g1  = (const float*)d_in[12];
    const float* b1  = (const float*)d_in[13];

    char* ws = (char*)d_ws;
    const size_t MB = 1024ull * 1024ull;
    float*          Qp   = (float*)(ws);                      // 32 MB (reused as Y)
    unsigned short* Kbf  = (unsigned short*)(ws + 32 * MB);   // 16 MB (K projection)
    unsigned short* Vt   = (unsigned short*)(ws + 48 * MB);   // 16 MB (V^T per head)
    float*          O1   = (float*)(ws + 64 * MB);            // 32 MB
    unsigned short* Qbf  = (unsigned short*)(ws + 96 * MB);   // 16 MB (input Q bf16)
    unsigned short* Kibf = (unsigned short*)(ws + 112 * MB);  // 16 MB (input K bf16)
    unsigned short* Wqt  = (unsigned short*)(ws + 128 * MB);  //  2 MB each
    unsigned short* Wkt  = (unsigned short*)(ws + 130 * MB);
    unsigned short* Wvt  = (unsigned short*)(ws + 132 * MB);
    unsigned short* Wot  = (unsigned short*)(ws + 134 * MB);
    unsigned short* O1bf = (unsigned short*)(ws + 136 * MB);  // 16 MB

    dim3 tg(32, 32);
    wtrans_kernel<<<tg, 256, 0, stream>>>(Wq, Wqt);
    wtrans_kernel<<<tg, 256, 0, stream>>>(Wk, Wkt);
    wtrans_kernel<<<tg, 256, 0, stream>>>(Wv, Wvt);
    wtrans_kernel<<<tg, 256, 0, stream>>>(Wo, Wot);

    cvt_kernel<<<4096, 256, 0, stream>>>(Q,   Qbf);
    cvt_kernel<<<4096, 256, 0, stream>>>(Kin, Kibf);

    dim3 gg(16, 64);   // (N/64, M/128), M = B*NQ = 8192
    gemm_kernel<<<gg, 256, 0, stream>>>(Qbf,  Wqt, bq, nullptr, (void*)Qp,  0);
    gemm_kernel<<<gg, 256, 0, stream>>>(Kibf, Wkt, bk, nullptr, (void*)Kbf, 1);
    gemm_kernel<<<gg, 256, 0, stream>>>(Kibf, Wvt, bv, nullptr, (void*)Vt,  2);

    attn_kernel<<<dim3(BB * H_, NQ_ / 128), 256, 0, stream>>>(Qp, Kbf, Vt, O1);

    ln_kernel<<<BB * NQ_, 256, 0, stream>>>(O1, O1, O1bf, g0, b0);

    gemm_kernel<<<gg, 256, 0, stream>>>(O1bf, Wot, bo, O1, (void*)Qp, 3);

    ln_kernel<<<BB * NQ_, 256, 0, stream>>>(Qp, (float*)d_out, nullptr, g1, b1);
}
